// SoftmaxCosineSim_46840913330646
// MI455X (gfx1250) — compile-verified, matrix operand order checked by reading, not run on hardware
//
#include <hip/hip_runtime.h>

// SimCLR contrastive softmax for MI455X (gfx1250, wave32, WMMA).
// out[4096, 16384] fp32. Strategy:
//  K1: L2-normalize z1/z2 rows -> f16 matrices in d_ws (2 MB).
//  K2: fused GEMM+softmax. Fixed-shift softmax (logits bounded by 10), two
//      passes (sum, then recompute+write) so the only HBM traffic is the
//      mandatory 256 MB output write; GEMM operands (2 MB) live in L2/WGP$.

typedef __attribute__((ext_vector_type(16))) _Float16 v16h;
typedef __attribute__((ext_vector_type(8)))  _Float16 v8h;
typedef __attribute__((ext_vector_type(4)))  _Float16 v4h;
typedef __attribute__((ext_vector_type(8)))  float    v8f;

#define BROWS 4096
#define DDIM  128
#define OUT_W (4 * BROWS)
#define NT    512          // 16-wide column tiles per part (2B / 16)
#define NWAVES 8
#define ROWS_PER_WG 32

// ---------------------------------------------------------------------------
// K1: one wave per row. lane loads float4, wave-reduce sum of squares,
// rsqrt, convert to f16, store 4 halves (b64).
// ---------------------------------------------------------------------------
__global__ __launch_bounds__(256) void simclr_normalize(
    const float* __restrict__ z1, const float* __restrict__ z2,
    _Float16* __restrict__ a16, _Float16* __restrict__ b16)
{
    const int gwave = (blockIdx.x * blockDim.x + threadIdx.x) >> 5; // [0, 8192)
    const int lane  = threadIdx.x & 31;
    const float*  src = (gwave < BROWS) ? z1 : z2;
    _Float16*     dst = (gwave < BROWS) ? a16 : b16;
    const int     row = gwave & (BROWS - 1);

    const float4 v = ((const float4*)(src + (size_t)row * DDIM))[lane];
    float ss = v.x * v.x + v.y * v.y + v.z * v.z + v.w * v.w;
#pragma unroll
    for (int m = 16; m >= 1; m >>= 1) ss += __shfl_xor(ss, m, 32);
    const float r = rsqrtf(fmaxf(ss, 1e-12f));

    v4h h;
    h[0] = (_Float16)(v.x * r);
    h[1] = (_Float16)(v.y * r);
    h[2] = (_Float16)(v.z * r);
    h[3] = (_Float16)(v.w * r);
    ((v4h*)(dst + (size_t)row * DDIM))[lane] = h;
}

// ---------------------------------------------------------------------------
// Load one 16x32 f16 fragment (A or B side; Gram product -> same layout).
// ISA 16-bit layout: lanes 0-15 = M, VGPR0-3 hold K 0..7 (lanes 0-15) /
// 8..15 (lanes 16-31), VGPR4-7 hold K 16..23 / 24..31.
// => per lane: halves [base+0..7] and [base+16..23], base = 8*(lane>>4)+32*kk.
// Two 16-byte contiguous loads per lane (global_load_b128).
// ---------------------------------------------------------------------------
__device__ __forceinline__ v16h load_frag16(const _Float16* __restrict__ M,
                                            int tilebase, int kk, int lane)
{
    const int m  = lane & 15;
    const int kh = lane >> 4;
    const _Float16* p = M + (size_t)(tilebase + m) * DDIM + kk * 32 + kh * 8;
    v8h lo = *(const v8h*)(p);
    v8h hi = *(const v8h*)(p + 16);
    v16h r;
#pragma unroll
    for (int i = 0; i < 8; ++i) { r[i] = lo[i]; r[i + 8] = hi[i]; }
    return r;
}

// ---------------------------------------------------------------------------
// K2: grid = (4096/32, 2 parts), block = 256 threads = 8 waves.
// Each wave holds A-fragments for 32 rows (2 WMMA row-tiles) and strides
// over the 512 column tiles. Pass 1 accumulates exp row sums; LDS+barrier
// combines across waves; pass 2 recomputes and writes normalized values.
// ---------------------------------------------------------------------------
__global__ __launch_bounds__(256) void simclr_gemm_softmax(
    const _Float16* __restrict__ a16, const _Float16* __restrict__ b16,
    float* __restrict__ out)
{
    __shared__ float rowsum[ROWS_PER_WG];

    const int lane    = threadIdx.x & 31;
    const int wave    = threadIdx.x >> 5;
    const int p       = blockIdx.y;               // 0 = part1 (a rows), 1 = part2 (b rows)
    const int rowbase = blockIdx.x * ROWS_PER_WG;

    const _Float16* X = p ? b16 : a16;            // row-side + "same" half
    const _Float16* Y = p ? a16 : b16;            // "cross" half

    if (threadIdx.x < ROWS_PER_WG) rowsum[threadIdx.x] = 0.0f;
    __syncthreads();

    // Resident A fragments: 2 row-tiles x 4 K-chunks (K=128) = 64 VGPRs.
    v16h aq[2][4];
#pragma unroll
    for (int s = 0; s < 2; ++s)
#pragma unroll
        for (int k = 0; k < 4; ++k)
            aq[s][k] = load_frag16(X, rowbase + s * 16, k, lane);

    const int kh   = lane >> 4;
    const int nloc = lane & 15;

    // ---------------- pass 1: row sums of exp(10*dot - 10) ----------------
    float acc[2][8];
#pragma unroll
    for (int s = 0; s < 2; ++s)
#pragma unroll
        for (int r = 0; r < 8; ++r) acc[s][r] = 0.0f;

    for (int ct = wave; ct < NT; ct += NWAVES) {
        const bool diag    = (ct >= NT / 2);
        const int  colbase = diag ? (ct - NT / 2) * 16 : ct * 16;
        const _Float16* S  = diag ? X : Y;
        v16h bq[4];
#pragma unroll
        for (int k = 0; k < 4; ++k) bq[k] = load_frag16(S, colbase, k, lane);
        const int j = colbase + nloc;
#pragma unroll
        for (int s = 0; s < 2; ++s) {
            v8f c = {};
#pragma unroll
            for (int k = 0; k < 4; ++k)
                c = __builtin_amdgcn_wmma_f32_16x16x32_f16(
                        false, aq[s][k], false, bq[k], (short)0, c, false, false);
            const int ibase = rowbase + s * 16 + kh * 8;
#pragma unroll
            for (int r = 0; r < 8; ++r) {
                float e = __expf((c[r] - 1.0f) * 10.0f);   // exp(logit - 10)
                e = (diag && (ibase + r == j)) ? 0.0f : e; // masked diagonal
                acc[s][r] += e;
            }
        }
    }

    // Reduce over the 16 lanes that share a row (xor masks stay in-group).
#pragma unroll
    for (int s = 0; s < 2; ++s)
#pragma unroll
        for (int r = 0; r < 8; ++r) {
            float v = acc[s][r];
            v += __shfl_xor(v, 1, 32);
            v += __shfl_xor(v, 2, 32);
            v += __shfl_xor(v, 4, 32);
            v += __shfl_xor(v, 8, 32);
            acc[s][r] = v;
        }
    if (nloc == 0) {
#pragma unroll
        for (int s = 0; s < 2; ++s)
#pragma unroll
            for (int r = 0; r < 8; ++r)
                atomicAdd(&rowsum[s * 16 + kh * 8 + r], acc[s][r]);
    }
    __syncthreads();

    float vinv[2][8];
#pragma unroll
    for (int s = 0; s < 2; ++s)
#pragma unroll
        for (int r = 0; r < 8; ++r)
            vinv[s][r] = 1.0f / rowsum[s * 16 + kh * 8 + r];

    // ---------------- pass 2: recompute, scale, store ----------------
    for (int ct = wave; ct < NT; ct += NWAVES) {
        const bool diag    = (ct >= NT / 2);
        const int  colbase = diag ? (ct - NT / 2) * 16 : ct * 16;
        const _Float16* S  = diag ? X : Y;
        v16h bq[4];
#pragma unroll
        for (int k = 0; k < 4; ++k) bq[k] = load_frag16(S, colbase, k, lane);
        const int j = colbase + nloc;
        // part-local column = ct*16 + nloc works for both halves (256*16 = 4096)
        const size_t ocol = (size_t)p * (2 * BROWS) + (size_t)ct * 16 + nloc;
#pragma unroll
        for (int s = 0; s < 2; ++s) {
            v8f c = {};
#pragma unroll
            for (int k = 0; k < 4; ++k)
                c = __builtin_amdgcn_wmma_f32_16x16x32_f16(
                        false, aq[s][k], false, bq[k], (short)0, c, false, false);
            const int ibase = rowbase + s * 16 + kh * 8;
#pragma unroll
            for (int r = 0; r < 8; ++r) {
                float e = __expf((c[r] - 1.0f) * 10.0f);
                e = (diag && (ibase + r == j)) ? 0.0f : e;
                out[(size_t)(ibase + r) * OUT_W + ocol] = e * vinv[s][r];
            }
        }
    }
}

// ---------------------------------------------------------------------------
extern "C" void kernel_launch(void* const* d_in, const int* in_sizes, int n_in,
                              void* d_out, int out_size, void* d_ws, size_t ws_size,
                              hipStream_t stream)
{
    (void)in_sizes; (void)n_in; (void)out_size; (void)ws_size;

    const float* z1 = (const float*)d_in[0];
    const float* z2 = (const float*)d_in[1];
    float* out = (float*)d_out;

    _Float16* a16 = (_Float16*)d_ws;                       // 4096*128 f16 = 1 MB
    _Float16* b16 = a16 + (size_t)BROWS * DDIM;            // +1 MB (ws use: 2 MB)

    // K1: 8192 rows, one wave each; 8 waves per 256-thread block.
    simclr_normalize<<<(2 * BROWS) / 8, 256, 0, stream>>>(z1, z2, a16, b16);

    // K2: 128 row-groups x 2 parts, 256 threads (8 waves).
    dim3 grid(BROWS / ROWS_PER_WG, 2);
    simclr_gemm_softmax<<<grid, 256, 0, stream>>>(a16, b16, out);
}
